// Attention_45861660787031
// MI455X (gfx1250) — matrix operand order repaired and checked by reading, not compiled, not run on hardware
//
#include <hip/hip_runtime.h>
#include <hip/hip_bf16.h>

// ---------------------------------------------------------------------------
// Types for CDNA5 WMMA (wave32): V_WMMA_F32_16X16X32_BF16
//   A: 16x32 bf16  (v16bf per lane, lanes 0-15 row M=lane: K=0..7,16..23;
//                   lanes 16-31 same row: K=8..15,24..31)
//   B: 32x16 bf16  (v16bf per lane, lane n = col; lanes 0-15 K=0..7,16..23;
//                   lanes 16-31 K=8..15,24..31, same order as A)
//   C/D: 16x16 f32 (v8f; VGPR r = row r (+8 for lanes 16-31), lane%16 = col)
// ---------------------------------------------------------------------------
typedef __attribute__((ext_vector_type(16))) __bf16 bf16x16;
typedef __attribute__((ext_vector_type(8)))  float  floatx8;
typedef __attribute__((ext_vector_type(4)))  unsigned int u32x4;
typedef __attribute__((ext_vector_type(8)))  int i32x8;
typedef __attribute__((ext_vector_type(4)))  int i32x4;

union Frag16 { bf16x16 v; uint4 u[2]; };
union Pack8  { uint4 u; __bf16 h[8]; };

#define WMMA_BF16(a, b, c)                                                    \
  __builtin_amdgcn_wmma_f32_16x16x32_bf16(false, (a), false, (b), (short)0,   \
                                          (c), false, false)

__device__ __forceinline__ floatx8 vzero8() {
  floatx8 v;
#pragma unroll
  for (int i = 0; i < 8; ++i) v[i] = 0.0f;
  return v;
}

// ---------------------------------------------------------------------------
// Tensor Data Mover: async 2D tile (bf16) global -> LDS with row padding.
// D# built per CDNA5 ISA:
//   group0: count=1 | lds_addr | global_addr[56:0] | type=2
//   group1: data_size=2B, pad ctrl, tensor/tile dims, 48-bit dim0 stride
// pad_interval code: 0=2,1=4,2=8,3=16,4=32,... DWORDs ; pad_amount code: n+1 DW
// ---------------------------------------------------------------------------
__device__ __forceinline__ void tdm_load_2d_bf16(
    unsigned lds_off, const void* gptr, unsigned tile_w_elems,
    unsigned tile_h_rows, unsigned long long row_stride_elems,
    unsigned pad_interval_code, unsigned pad_amount_code) {
  unsigned long long ga = (unsigned long long)reinterpret_cast<uintptr_t>(gptr);
  u32x4 g0;
  g0[0] = 1u;                                   // count=1 (valid), user mode
  g0[1] = lds_off;                              // lds_addr (byte offset)
  g0[2] = (unsigned)(ga & 0xffffffffu);         // global_addr[31:0]
  g0[3] = (unsigned)(ga >> 32) | (2u << 30);    // global_addr[56:32] | type=2

  i32x8 g1;
  g1[0] = (int)((1u << 16) |                    // data_size = 2 bytes
                (1u << 20) |                    // pad_enable
                (pad_interval_code << 22) | (pad_amount_code << 25));
  g1[1] = (int)(tile_w_elems << 16);            // tensor_dim0[15:0]
  g1[2] = (int)((tile_w_elems >> 16) | (tile_h_rows << 16));  // d0 hi|d1 lo
  g1[3] = (int)((tile_h_rows >> 16) | (tile_w_elems << 16));  // d1 hi|tile_dim0
  g1[4] = (int)tile_h_rows;                     // tile_dim1 (tile_dim2 = 0)
  g1[5] = (int)(unsigned)(row_stride_elems & 0xffffffffu);    // d0 stride lo
  g1[6] = (int)(unsigned)(row_stride_elems >> 32);            // d0 stride hi
  g1[7] = 0;

  i32x4 z4 = {0, 0, 0, 0};
#if defined(__clang_major__) && __clang_major__ >= 23
  i32x8 z8 = {0, 0, 0, 0, 0, 0, 0, 0};
  __builtin_amdgcn_tensor_load_to_lds(g0, g1, z4, z4, z8, 0);
#else
  __builtin_amdgcn_tensor_load_to_lds(g0, g1, z4, z4, 0);
#endif
}

// Problem constants
static constexpr int Bn = 8;      // batch
static constexpr int Sq = 1024;   // seq len
static constexpr int Dm = 768;    // model dim
static constexpr int Hn = 12;     // heads
static constexpr int Dh = 64;     // head dim
static constexpr int Jq = 3 * Dm; // 2304 (packed Q|K|V columns)

// ---------------------------------------------------------------------------
// fp32 -> bf16 elementwise convert
// ---------------------------------------------------------------------------
__global__ void k_cvt_bf16(const float* __restrict__ in, void* __restrict__ outv,
                           int n) {
  __bf16* out = (__bf16*)outv;
  int i = blockIdx.x * blockDim.x + threadIdx.x;
  if (i < n) out[i] = (__bf16)in[i];
}

// ---------------------------------------------------------------------------
// Pack W_Q/W_K/W_V [H][D][Dh] fp32 -> Wqkv bf16 [D][3*Dm], col j =
// which*768 + h*64 + e ; also pack biases into bqkv[2304].
// ---------------------------------------------------------------------------
__global__ void k_pack_wqkv(const float* __restrict__ WQ,
                            const float* __restrict__ WK,
                            const float* __restrict__ WV,
                            const float* __restrict__ bQ,
                            const float* __restrict__ bK,
                            const float* __restrict__ bV,
                            void* __restrict__ Wpv,
                            float* __restrict__ bp) {
  __bf16* Wp = (__bf16*)Wpv;
  int idx = blockIdx.x * blockDim.x + threadIdx.x;
  if (idx < Dm * Jq) {
    int d = idx / Jq;
    int j = idx % Jq;
    int which = j / Dm;
    int hj = j % Dm;
    int h = hj / Dh, e = hj % Dh;
    const float* W = (which == 0) ? WQ : (which == 1) ? WK : WV;
    Wp[idx] = (__bf16)W[((size_t)h * Dm + d) * Dh + e];
  }
  if (idx < Jq) {
    int which = idx / Dm;
    int hj = idx % Dm;
    const float* bb = (which == 0) ? bQ : (which == 1) ? bK : bV;
    bp[idx] = bb[hj];
  }
}

// ---------------------------------------------------------------------------
// bf16 GEMM: C[M][N] = A[M][K] * B[K][N] + bias[N]
// Block tile 256x64, BK=32, 256 threads (8 waves; wave = 32 rows x 64 cols).
// A tile staged by TDM (async tensor DMA, padded rows); B tile transposed
// manually into LDS [n][k] so B fragments are contiguous ds_load_b128 pairs.
// ---------------------------------------------------------------------------
template <bool BF16OUT>
__global__ __launch_bounds__(256) void k_gemm_bf16(
    const void* __restrict__ Av, const void* __restrict__ Bv,
    const float* __restrict__ bias, void* __restrict__ Cout, int M, int N,
    int K) {
  const __bf16* A = (const __bf16*)Av;
  const __bf16* B = (const __bf16*)Bv;

  __shared__ __align__(16) __bf16 As[256][40];  // [row][k], 80B stride (64+16)
  __shared__ __align__(16) __bf16 Bs[64][40];   // transposed [n][k]

  const int tid = threadIdx.x;
  const int wave = tid >> 5;
  const int lane = tid & 31;
  const int lhalf = lane >> 4;
  const int lmod = lane & 15;
  const int m0 = blockIdx.x * 256;
  const int n0 = blockIdx.y * 64;
  const unsigned asOff =
      (unsigned)reinterpret_cast<uintptr_t>(&As[0][0]);

  floatx8 acc[2][4];
#pragma unroll
  for (int g = 0; g < 2; ++g)
#pragma unroll
    for (int f = 0; f < 4; ++f) acc[g][f] = vzero8();

  for (int kt = 0; kt < K; kt += 32) {
    // A tile 256x32 via Tensor Data Mover (64B rows + 16B pad -> 80B stride)
    if (wave == 0)
      tdm_load_2d_bf16(asOff, A + (size_t)m0 * K + kt, 32u, 256u,
                       (unsigned long long)K, /*interval 16DW*/ 3u,
                       /*pad 4DW*/ 3u);
    // B tile: 32x64 bf16 loaded row-major, stored transposed [n][k]
    {
      int k = tid >> 3;
      int c = (tid & 7) * 8;
      Pack8 p;
      p.u = *(const uint4*)(B + (size_t)(kt + k) * N + n0 + c);
#pragma unroll
      for (int j = 0; j < 8; ++j) Bs[c + j][k] = p.h[j];
    }
    if (wave == 0) __builtin_amdgcn_s_wait_tensorcnt(0);
    __syncthreads();

    Frag16 fa[2];
#pragma unroll
    for (int g = 0; g < 2; ++g) {
      int row = wave * 32 + g * 16 + lmod;
      fa[g].u[0] = *(const uint4*)&As[row][lhalf * 8];
      fa[g].u[1] = *(const uint4*)&As[row][lhalf * 8 + 16];
    }
#pragma unroll
    for (int f = 0; f < 4; ++f) {
      Frag16 fb;
      fb.u[0] = *(const uint4*)&Bs[f * 16 + lmod][lhalf * 8];
      fb.u[1] = *(const uint4*)&Bs[f * 16 + lmod][lhalf * 8 + 16];
#pragma unroll
      for (int g = 0; g < 2; ++g) acc[g][f] = WMMA_BF16(fa[g].v, fb.v, acc[g][f]);
    }
    __syncthreads();
  }

#pragma unroll
  for (int g = 0; g < 2; ++g)
#pragma unroll
    for (int f = 0; f < 4; ++f)
#pragma unroll
      for (int r = 0; r < 8; ++r) {
        int row = m0 + wave * 32 + g * 16 + r + 8 * lhalf;
        int col = n0 + f * 16 + lmod;
        float v = acc[g][f][r] + bias[col];
        if (BF16OUT)
          ((__bf16*)Cout)[(size_t)row * N + col] = (__bf16)v;
        else
          ((float*)Cout)[(size_t)row * N + col] = v;
      }
}

// ---------------------------------------------------------------------------
// Causal flash attention. Grid: B*H*(S/64) blocks, 128 threads (4 waves).
// Each wave owns a 16-row Q tile; 32-key K/V tiles shared via LDS.
// K tile staged by TDM ([key][e] == K^T B-fragment layout, padded rows);
// V tile transposed manually into [e][key] for the P*V B-fragments.
// ---------------------------------------------------------------------------
__global__ __launch_bounds__(128) void k_attn(const void* __restrict__ QKVv,
                                              void* __restrict__ Zv) {
  const __bf16* QKV = (const __bf16*)QKVv;
  __bf16* Z = (__bf16*)Zv;

  const int tid = threadIdx.x;
  const int wave = tid >> 5;
  const int lane = tid & 31;
  const int lhalf = lane >> 4;
  const int lmod = lane & 15;

  const int qb = blockIdx.x & 15;  // S/64 = 16 q-blocks
  const int bh = blockIdx.x >> 4;
  const int h = bh % Hn;
  const int b = bh / Hn;
  const int q0 = qb * 64;
  const int qt = q0 + wave * 16;

  __shared__ __align__(16) __bf16 Ks[32][72];     // [key][e], 144B stride
  __shared__ __align__(16) __bf16 Vt[64][40];     // [e][key]
  __shared__ __align__(16) __bf16 Ps[4][16][40];  // per-wave P staging
  const unsigned ksOff = (unsigned)reinterpret_cast<uintptr_t>(&Ks[0][0]);

  // Preload resident Q A-fragments (e-chunks 0..31 and 32..63)
  Frag16 qa[2];
  {
    const __bf16* qptr = QKV + (size_t)(b * Sq + qt + lmod) * Jq + h * Dh;
#pragma unroll
    for (int i = 0; i < 2; ++i) {
      qa[i].u[0] = *(const uint4*)(qptr + i * 32 + lhalf * 8);
      qa[i].u[1] = *(const uint4*)(qptr + i * 32 + lhalf * 8 + 16);
    }
  }

  floatx8 zacc[4];
#pragma unroll
  for (int f = 0; f < 4; ++f) zacc[f] = vzero8();
  float m[8], l[8];
#pragma unroll
  for (int r = 0; r < 8; ++r) { m[r] = -1e30f; l[r] = 0.0f; }

  const float SC = 0.125f * 1.44269504088896341f;  // 1/sqrt(64) * log2(e)
  const int kend = q0 + 63;

  for (int kk = 0; kk <= kend; kk += 32) {
    // --- K tile 32x64 via TDM (128B rows + 16B pad -> 144B stride) ---
    if (wave == 0)
      tdm_load_2d_bf16(ksOff,
                       QKV + (size_t)(b * Sq + kk) * Jq + Dm + h * Dh, 64u, 32u,
                       (unsigned long long)Jq, /*interval 32DW*/ 4u,
                       /*pad 4DW*/ 3u);
    // --- V tile load (transposed into Vt[e][key]) by all threads ---
#pragma unroll
    for (int it = 0; it < 2; ++it) {
      int idx = tid + it * 128;
      int key = idx >> 3;
      int c = (idx & 7) * 8;
      Pack8 p;
      p.u = *(const uint4*)(QKV + (size_t)(b * Sq + kk + key) * Jq + 2 * Dm +
                            h * Dh + c);
#pragma unroll
      for (int j = 0; j < 8; ++j) Vt[c + j][key] = p.h[j];
    }
    if (wave == 0) __builtin_amdgcn_s_wait_tensorcnt(0);
    __syncthreads();

    if (kk <= qt + 15) {  // wave-uniform: skip fully-masked key blocks
      // --- scores: Q(16x64) * K^T(64x32) -> two 16x16 C frags ---
      floatx8 c0 = vzero8(), c1 = vzero8();
#pragma unroll
      for (int ei = 0; ei < 2; ++ei) {
        Frag16 kb0, kb1;
        kb0.u[0] = *(const uint4*)&Ks[lmod][ei * 32 + lhalf * 8];
        kb0.u[1] = *(const uint4*)&Ks[lmod][ei * 32 + lhalf * 8 + 16];
        kb1.u[0] = *(const uint4*)&Ks[16 + lmod][ei * 32 + lhalf * 8];
        kb1.u[1] = *(const uint4*)&Ks[16 + lmod][ei * 32 + lhalf * 8 + 16];
        c0 = WMMA_BF16(qa[ei].v, kb0.v, c0);
        c1 = WMMA_BF16(qa[ei].v, kb1.v, c1);
      }

      // --- online softmax in C-fragment layout ---
      float p0a[8], p1a[8];
#pragma unroll
      for (int r = 0; r < 8; ++r) {
        int q = qt + r + 8 * lhalf;
        float s0 = ((kk + lmod) > q ? -1e5f : c0[r]) * SC;
        float s1 = ((kk + 16 + lmod) > q ? -1e5f : c1[r]) * SC;
        float rmax = fmaxf(s0, s1);
#pragma unroll
        for (int off = 1; off < 16; off <<= 1)
          rmax = fmaxf(rmax, __shfl_xor(rmax, off, 32));
        float mn = fmaxf(m[r], rmax);
        float alpha = exp2f(m[r] - mn);
        m[r] = mn;
        float p0 = exp2f(s0 - mn);
        float p1 = exp2f(s1 - mn);
        float rs = p0 + p1;
#pragma unroll
        for (int off = 1; off < 16; off <<= 1) rs += __shfl_xor(rs, off, 32);
        l[r] = l[r] * alpha + rs;
#pragma unroll
        for (int f = 0; f < 4; ++f) zacc[f][r] *= alpha;
        p0a[r] = p0;
        p1a[r] = p1;
      }

      // --- C-layout P -> A-layout via wave-private LDS ---
#pragma unroll
      for (int r = 0; r < 8; ++r) {
        int pr = r + 8 * lhalf;
        Ps[wave][pr][lmod] = (__bf16)p0a[r];
        Ps[wave][pr][16 + lmod] = (__bf16)p1a[r];
      }
      asm volatile("s_wait_dscnt 0" ::: "memory");  // wave-local RAW fence
      Frag16 pa;
      pa.u[0] = *(const uint4*)&Ps[wave][lmod][lhalf * 8];
      pa.u[1] = *(const uint4*)&Ps[wave][lmod][lhalf * 8 + 16];

      // --- z += P(16x32) * V(32x64): 4 N-tiles ---
#pragma unroll
      for (int f = 0; f < 4; ++f) {
        Frag16 vb;
        vb.u[0] = *(const uint4*)&Vt[f * 16 + lmod][lhalf * 8];
        vb.u[1] = *(const uint4*)&Vt[f * 16 + lmod][lhalf * 8 + 16];
        zacc[f] = WMMA_BF16(pa.v, vb.v, zacc[f]);
      }
    }
    __syncthreads();
  }

  // --- normalize and store Z (bf16) ---
  float inv[8];
#pragma unroll
  for (int r = 0; r < 8; ++r) inv[r] = 1.0f / l[r];
#pragma unroll
  for (int f = 0; f < 4; ++f)
#pragma unroll
    for (int r = 0; r < 8; ++r) {
      int q = qt + r + 8 * lhalf;
      int e = f * 16 + lmod;
      Z[(size_t)(b * Sq + q) * Dm + h * Dh + e] = (__bf16)(zacc[f][r] * inv[r]);
    }
}

// ---------------------------------------------------------------------------
// Host-side launcher
// ---------------------------------------------------------------------------
extern "C" void kernel_launch(void* const* d_in, const int* in_sizes, int n_in,
                              void* d_out, int out_size, void* d_ws,
                              size_t ws_size, hipStream_t stream) {
  (void)in_sizes; (void)n_in; (void)out_size; (void)ws_size;

  const float* x  = (const float*)d_in[0];
  const float* WQ = (const float*)d_in[1];
  const float* WK = (const float*)d_in[2];
  const float* WV = (const float*)d_in[3];
  const float* WO = (const float*)d_in[4];
  const float* bQ = (const float*)d_in[5];
  const float* bK = (const float*)d_in[6];
  const float* bV = (const float*)d_in[7];
  const float* bO = (const float*)d_in[8];

  const int tokens = Bn * Sq;  // 8192

  char* ws = (char*)d_ws;
  size_t off = 0;
  auto carve = [&](size_t bytes) {
    void* p = ws + off;
    off += (bytes + 255) & ~(size_t)255;
    return p;
  };
  void* Xb   = carve((size_t)tokens * Dm * 2);  // x in bf16
  void* Wqkv = carve((size_t)Dm * Jq * 2);      // packed QKV weights bf16
  float* bqkv = (float*)carve((size_t)Jq * 4);  // packed QKV bias fp32
  void* Wob  = carve((size_t)Dm * Dm * 2);      // W_O bf16 [h*64+e][d]
  void* QKV  = carve((size_t)tokens * Jq * 2);  // Q|K|V bf16
  void* Zb   = carve((size_t)tokens * Dm * 2);  // attn output bf16

  // 1) precision packing
  k_cvt_bf16<<<(tokens * Dm + 255) / 256, 256, 0, stream>>>(x, Xb, tokens * Dm);
  k_cvt_bf16<<<(Dm * Dm + 255) / 256, 256, 0, stream>>>(WO, Wob, Dm * Dm);
  k_pack_wqkv<<<(Dm * Jq + 255) / 256, 256, 0, stream>>>(WQ, WK, WV, bQ, bK, bV,
                                                         Wqkv, bqkv);
  // 2) fused QKV projection: [8192x768] x [768x2304]
  k_gemm_bf16<true><<<dim3(tokens / 256, Jq / 64), 256, 0, stream>>>(
      Xb, Wqkv, bqkv, QKV, tokens, Jq, Dm);
  // 3) causal flash attention
  k_attn<<<Bn * Hn * (Sq / 64), 128, 0, stream>>>(QKV, Zb);
  // 4) output projection: [8192x768] x [768x768] + b_O -> fp32 out
  k_gemm_bf16<false><<<dim3(tokens / 256, Dm / 64), 256, 0, stream>>>(
      Zb, Wob, bO, d_out, tokens, Dm, Dm);
}
